// EncoderBlock_5978594476399
// MI455X (gfx1250) — compile-verified, matrix operand order checked by reading, not run on hardware
//
#include <hip/hip_runtime.h>
#include <hip/hip_bf16.h>
#include <math.h>

// ---------------------------------------------------------------------------
// EncoderBlock for MI455X (gfx1250).
//   D=1024, H=16, HD=64, FF=4096, B=2, S=2048.
// f16 WMMA (v_wmma_f32_16x16x32_f16) for all GEMMs + flash attention,
// double-buffered TDM (tensor_load_to_lds) staging of GEMM tiles into LDS.
// ---------------------------------------------------------------------------

typedef __attribute__((ext_vector_type(16))) _Float16 v16h;
typedef __attribute__((ext_vector_type(8)))  _Float16 v8h;
typedef __attribute__((ext_vector_type(8)))  float    v8f;
typedef __attribute__((ext_vector_type(4)))  unsigned int u32x4;
typedef __attribute__((ext_vector_type(8)))  int      i32x8;
typedef __attribute__((ext_vector_type(4)))  int      i32x4;

#define D_MODEL 1024
#define NHEAD   16
#define HD      64
#define FF_DIM  4096
#define SEQ     2048
#define BATCH   2
#define ROWS    (BATCH * SEQ)   /* 4096 */
#define LN_EPS  1e-6f

#if defined(__has_builtin)
#if __has_builtin(__builtin_amdgcn_tensor_load_to_lds)
#define HAVE_TDM 1
#endif
#endif

// ---- WMMA helpers ---------------------------------------------------------

static __device__ __forceinline__ v8f wmma16x16x32(v16h a, v16h b, v8f c) {
  return __builtin_amdgcn_wmma_f32_16x16x32_f16(false, a, false, b, (short)0, c,
                                                false, false);
}

// A/B fragment loader from a row-major [16 x >=32] tile (row = M for A, N for B;
// K contiguous within the row). ISA 16-bit A layout:
//   lanes 0-15:  M=lane,    elems 0..7 -> K=0..7,  elems 8..15 -> K=16..23
//   lanes 16-31: M=lane-16, elems 0..7 -> K=8..15, elems 8..15 -> K=24..31
static __device__ __forceinline__ v16h load_frag_rowmajor(const _Float16* base,
                                                          int ld, int lane) {
  const int r  = lane & 15;
  const int kb = (lane >> 4) * 8;
  const _Float16* p = base + (size_t)r * ld + kb;
  v16h f;
#pragma unroll
  for (int i = 0; i < 8; ++i) f[i] = p[i];
#pragma unroll
  for (int i = 0; i < 8; ++i) f[8 + i] = p[16 + i];
  return f;
}

// ---- TDM: DMA a [rows=128 x 32] f16 tile (row stride = K elems) to LDS ----

#if HAVE_TDM
static __device__ __forceinline__ void tdm_load_tile128x32(
    const _Float16* gbase, unsigned int ldsOff, int K) {
  const unsigned long long ga = (unsigned long long)(uintptr_t)gbase;
  u32x4 g0;
  g0[0] = 1u;                                       // count=1, user mode
  g0[1] = ldsOff;                                   // lds_addr (bytes)
  g0[2] = (unsigned int)ga;                         // global_addr[31:0]
  g0[3] = (unsigned int)((ga >> 32) & 0x01FFFFFFu)  // global_addr[56:32]
          | (2u << 30);                             // type = 2 ("image")
  i32x8 g1;
  const unsigned int td0 = (unsigned int)K;         // tensor_dim0 (elements)
  g1[0] = (int)(1u << 16);                          // data_size = 1 -> 2 bytes
  g1[1] = (int)((td0 & 0xFFFFu) << 16);             // tensor_dim0[15:0] @ 63:48
  g1[2] = (int)((td0 >> 16) | (128u << 16));        // dim0[31:16] | tensor_dim1=128
  g1[3] = (int)(32u << 16);                         // tensor_dim1[31:16]=0 | tile_dim0=32
  g1[4] = (int)128;                                 // tile_dim1=128, tile_dim2=0
  g1[5] = (int)td0;                                 // tensor_dim0_stride[31:0]
  g1[6] = 0;                                        // stride[47:32]=0, dim1_stride lo
  g1[7] = 0;
  i32x4 z4 = {0, 0, 0, 0};
  i32x8 z8 = {0, 0, 0, 0, 0, 0, 0, 0};
  __builtin_amdgcn_tensor_load_to_lds(g0, g1, z4, z4, z8, 0);
}
#endif

// ---- elementwise f32 -> f16 ----------------------------------------------

__global__ void cvt_kernel(const float* __restrict__ in,
                           _Float16* __restrict__ out, int n) {
  int i = blockIdx.x * blockDim.x + threadIdx.x;
  if (i < n) out[i] = (_Float16)in[i];
}

// ---- additive attention mask: 0 or -1e9 -----------------------------------

__global__ void maskf_kernel(const int* __restrict__ mask,
                             float* __restrict__ madd, int n) {
  int i = blockIdx.x * blockDim.x + threadIdx.x;
  if (i < n) madd[i] = (mask[i] != 0) ? 0.f : -1e9f;
}

// ---- V transpose to head-major vt[B,H,HD,SEQ] (f16) -----------------------

__global__ void vtrans_kernel(const _Float16* __restrict__ vsrc,
                              _Float16* __restrict__ vt) {
  const size_t o = (size_t)blockIdx.x * 256 + threadIdx.x;  // output index
  const int s   = (int)(o & (SEQ - 1));
  const size_t r = o >> 11;                 // (b*H + h)*HD + d
  const int d   = (int)(r & (HD - 1));
  const int hh  = (int)((r >> 6) & (NHEAD - 1));
  const int b   = (int)(r >> 10);
  vt[o] = vsrc[((size_t)(b * SEQ + s)) * D_MODEL + hh * HD + d];
}

// ---- LayerNorm (unbiased std, eps added to sd) -> f16 ---------------------

__global__ __launch_bounds__(256)
void ln_kernel(const float* __restrict__ x, const float* __restrict__ alpha,
               const float* __restrict__ bias, _Float16* __restrict__ out) {
  __shared__ float red[256];
  const int row = blockIdx.x;
  const int t = threadIdx.x;
  const float* xr = x + (size_t)row * D_MODEL;
  float v[4];
  float s = 0.f;
#pragma unroll
  for (int i = 0; i < 4; ++i) { v[i] = xr[t + i * 256]; s += v[i]; }
  red[t] = s; __syncthreads();
  for (int off = 128; off > 0; off >>= 1) {
    if (t < off) red[t] += red[t + off];
    __syncthreads();
  }
  const float mu = red[0] * (1.f / D_MODEL);
  __syncthreads();
  float ss = 0.f;
#pragma unroll
  for (int i = 0; i < 4; ++i) { float d = v[i] - mu; ss += d * d; }
  red[t] = ss; __syncthreads();
  for (int off = 128; off > 0; off >>= 1) {
    if (t < off) red[t] += red[t + off];
    __syncthreads();
  }
  const float sd  = sqrtf(red[0] / (float)(D_MODEL - 1));
  const float inv = 1.f / (sd + LN_EPS);
#pragma unroll
  for (int i = 0; i < 4; ++i) {
    int c = t + i * 256;
    out[(size_t)row * D_MODEL + c] =
        (_Float16)(alpha[c] * (v[i] - mu) * inv + bias[c]);
  }
}

// ---- Tiled WMMA GEMM:  C[M,N] = act(A[M,K] * W[N,K]^T + bias) (+resid) ----
// Block = 256 threads (8 waves), tile 128x128, BK=32. Waves: 2 (M) x 4 (N);
// each wave owns 64x32 of output (4 M-frags x 2 N-frags = 8 WMMA / k-step).
// TDM path: double-buffered LDS; wave 0 issues tile k+1's DMA before waiting
// tensorcnt<=2 (per-wave tensor ops complete in order -> tile k has landed),
// hiding HBM latency under the current tile's WMMAs.

template <bool RELU, bool RESID, bool BIAS, bool OUTF32, bool OUTF16>
__global__ __launch_bounds__(256)
void gemm_kernel(const _Float16* __restrict__ A, const _Float16* __restrict__ W,
                 const float* __restrict__ bias, const float* __restrict__ resid,
                 float* __restrict__ outf, _Float16* __restrict__ outh,
                 int M, int N, int K) {
#if HAVE_TDM
  __shared__ _Float16 As[2][128 * 32];
  __shared__ _Float16 Bs[2][128 * 32];
#else
  __shared__ _Float16 As[1][128 * 32];
  __shared__ _Float16 Bs[1][128 * 32];
#endif
  const int t    = threadIdx.x;
  const int lane = t & 31;
  const int wave = t >> 5;
  const int wm   = wave >> 2;       // 0..1  (64 rows of M each)
  const int wn   = wave & 3;        // 0..3  (32 cols of N each)
  const int m0   = blockIdx.y * 128;
  const int n0   = blockIdx.x * 128;

  v8f acc[4][2];
#pragma unroll
  for (int i = 0; i < 4; ++i)
#pragma unroll
    for (int j = 0; j < 2; ++j) acc[i][j] = v8f{};

  auto compute_tile = [&](const _Float16* Ab, const _Float16* Bb) {
    v16h bf[2], af[4];
#pragma unroll
    for (int nf = 0; nf < 2; ++nf)
      bf[nf] = load_frag_rowmajor(&Bb[(wn * 32 + nf * 16) * 32], 32, lane);
#pragma unroll
    for (int mf = 0; mf < 4; ++mf)
      af[mf] = load_frag_rowmajor(&Ab[(wm * 64 + mf * 16) * 32], 32, lane);
#pragma unroll
    for (int mf = 0; mf < 4; ++mf)
#pragma unroll
      for (int nf = 0; nf < 2; ++nf)
        acc[mf][nf] = wmma16x16x32(af[mf], bf[nf], acc[mf][nf]);
  };

#if HAVE_TDM
  const unsigned int ldsA[2] = {(unsigned int)(uintptr_t)(void*)&As[0][0],
                                (unsigned int)(uintptr_t)(void*)&As[1][0]};
  const unsigned int ldsB[2] = {(unsigned int)(uintptr_t)(void*)&Bs[0][0],
                                (unsigned int)(uintptr_t)(void*)&Bs[1][0]};
  const _Float16* Agb = A + (size_t)m0 * K;
  const _Float16* Wgb = W + (size_t)n0 * K;
  const int nk = K >> 5;

  if (wave == 0) {                       // prologue: tile 0 into buffer 0
    tdm_load_tile128x32(Agb, ldsA[0], K);
    tdm_load_tile128x32(Wgb, ldsB[0], K);
  }
  for (int ki = 0; ki < nk; ++ki) {
    const int cur = ki & 1;
    if (wave == 0) {
      if (ki + 1 < nk) {                 // prefetch next tile into other buffer
        const int kn = (ki + 1) << 5;
        tdm_load_tile128x32(Agb + kn, ldsA[cur ^ 1], K);
        tdm_load_tile128x32(Wgb + kn, ldsB[cur ^ 1], K);
        __builtin_amdgcn_s_wait_tensorcnt(2);  // tile ki's pair has landed
      } else {
        __builtin_amdgcn_s_wait_tensorcnt(0);
      }
    }
    __syncthreads();                     // buffer `cur` ready for all waves
    compute_tile(&As[cur][0], &Bs[cur][0]);
    __syncthreads();                     // all waves done before reuse of buffer
  }
#else
  const int ldRow = t >> 1;              // 0..127
  const int ldCol = (t & 1) * 16;        // 0 or 16
  for (int kt = 0; kt < K; kt += 32) {
    __syncthreads();
    *reinterpret_cast<v8h*>(&As[0][ldRow * 32 + ldCol]) =
        *reinterpret_cast<const v8h*>(A + (size_t)(m0 + ldRow) * K + kt + ldCol);
    *reinterpret_cast<v8h*>(&As[0][ldRow * 32 + ldCol + 8]) =
        *reinterpret_cast<const v8h*>(A + (size_t)(m0 + ldRow) * K + kt + ldCol + 8);
    *reinterpret_cast<v8h*>(&Bs[0][ldRow * 32 + ldCol]) =
        *reinterpret_cast<const v8h*>(W + (size_t)(n0 + ldRow) * K + kt + ldCol);
    *reinterpret_cast<v8h*>(&Bs[0][ldRow * 32 + ldCol + 8]) =
        *reinterpret_cast<const v8h*>(W + (size_t)(n0 + ldRow) * K + kt + ldCol + 8);
    __syncthreads();
    compute_tile(&As[0][0], &Bs[0][0]);
  }
#endif

  const int hi8 = (lane >> 4) * 8;
#pragma unroll
  for (int nf = 0; nf < 2; ++nf) {
    const int gn = n0 + wn * 32 + nf * 16 + (lane & 15);
    float bv = 0.f;
    if constexpr (BIAS) bv = bias[gn];
#pragma unroll
    for (int mf = 0; mf < 4; ++mf) {
#pragma unroll
      for (int v = 0; v < 8; ++v) {
        const int gm = m0 + wm * 64 + mf * 16 + hi8 + v;  // D layout: M = v + hi8
        float val = acc[mf][nf][v];
        if constexpr (BIAS)  val += bv;
        if constexpr (RELU)  val = fmaxf(val, 0.f);
        const size_t idx = (size_t)gm * N + gn;
        if constexpr (RESID) val += resid[idx];
        if constexpr (OUTF32) outf[idx] = val;
        if constexpr (OUTF16) outh[idx] = (_Float16)val;
      }
    }
  }
}

// ---- Flash attention, 1 wave per (b, h, 16 query rows) --------------------
// Computes S^T = K.Q^T so the D-fragment (q = lane&15, key = v + hi8) matches
// the A-operand layout required for P.V with no cross-lane transpose.
// V is pre-transposed to vt[B,H,HD,SEQ] so its B-fragments are contiguous.

__global__ __launch_bounds__(32)
void attn_kernel(const _Float16* __restrict__ qh, const _Float16* __restrict__ kh,
                 const _Float16* __restrict__ vt, const float* __restrict__ madd,
                 _Float16* __restrict__ ctxh) {
  const int lane = threadIdx.x;
  const int q0   = blockIdx.x * 16;
  const int h    = blockIdx.y;
  const int b    = blockIdx.z;
  const size_t hoff = (size_t)h * HD;
  const int hi8  = (lane >> 4) * 8;
  const float scale = 0.125f;  // 1/sqrt(HD)

  const _Float16* qbase = qh + ((size_t)(b * SEQ + q0)) * D_MODEL + hoff;
  v16h qb_lo = load_frag_rowmajor(qbase,      D_MODEL, lane);   // B: N=q, K=hd
  v16h qb_hi = load_frag_rowmajor(qbase + 32, D_MODEL, lane);

  v8f ctx[4] = {v8f{}, v8f{}, v8f{}, v8f{}};
  float m = -1e30f, l = 0.f;
  const float* mrow = madd + (size_t)b * SEQ;
  const _Float16* vtb = vt + ((size_t)(b * NHEAD + h) * HD) * SEQ;

  for (int j = 0; j < SEQ; j += 32) {
    const _Float16* kbase = kh + ((size_t)(b * SEQ + j)) * D_MODEL + hoff;
    v16h ka0l = load_frag_rowmajor(kbase,                     D_MODEL, lane);
    v16h ka0h = load_frag_rowmajor(kbase + 32,                D_MODEL, lane);
    v16h ka1l = load_frag_rowmajor(kbase + 16 * D_MODEL,      D_MODEL, lane);
    v16h ka1h = load_frag_rowmajor(kbase + 16 * D_MODEL + 32, D_MODEL, lane);
    v8f s0 = {}; s0 = wmma16x16x32(ka0l, qb_lo, s0); s0 = wmma16x16x32(ka0h, qb_hi, s0);
    v8f s1 = {}; s1 = wmma16x16x32(ka1l, qb_lo, s1); s1 = wmma16x16x32(ka1h, qb_hi, s1);

    // per-lane: q = lane&15 ; element v: key = j + hi8 + v (s0) / +16 (s1)
    float sv0[8], sv1[8];
    float tmax = -1e30f;
#pragma unroll
    for (int v = 0; v < 8; ++v) {
      const int key0 = j + hi8 + v;
      const float a = fmaf(s0[v], scale, mrow[key0]);
      const float c = fmaf(s1[v], scale, mrow[key0 + 16]);
      sv0[v] = a; sv1[v] = c;
      tmax = fmaxf(tmax, fmaxf(a, c));
    }
    tmax = fmaxf(tmax, __shfl_xor(tmax, 16, 32));
    const float mnew  = fmaxf(m, tmax);
    const float alpha = __expf(m - mnew);
    float psum = 0.f;
    v16h pa;
#pragma unroll
    for (int v = 0; v < 8; ++v) {
      const float p0 = __expf(sv0[v] - mnew);
      const float p1 = __expf(sv1[v] - mnew);
      pa[v]     = (_Float16)p0;    // A-frag elem e<8  -> key kb+e
      pa[8 + v] = (_Float16)p1;    // A-frag elem e>=8 -> key kb+16+(e-8)
      psum += p0 + p1;
    }
    psum += __shfl_xor(psum, 16, 32);
    l = l * alpha + psum;
    m = mnew;

    // rescale ctx: row q = v + hi8 -> alpha lives in lane (v + hi8)
    float rs[8];
#pragma unroll
    for (int v = 0; v < 8; ++v) rs[v] = __shfl(alpha, hi8 + v, 32);
#pragma unroll
    for (int nf = 0; nf < 4; ++nf)
#pragma unroll
      for (int v = 0; v < 8; ++v) ctx[nf][v] *= rs[v];

#pragma unroll
    for (int nf = 0; nf < 4; ++nf) {
      // B operand: N = hd (rows of vt), K = keys (contiguous)
      v16h vb = load_frag_rowmajor(vtb + (size_t)(nf * 16) * SEQ + j, SEQ, lane);
      ctx[nf] = wmma16x16x32(pa, vb, ctx[nf]);
    }
  }

  const float linv = 1.f / l;
  float rs[8];
#pragma unroll
  for (int v = 0; v < 8; ++v) rs[v] = __shfl(linv, hi8 + v, 32);
  _Float16* obase = ctxh + ((size_t)(b * SEQ + q0)) * D_MODEL + hoff;
#pragma unroll
  for (int nf = 0; nf < 4; ++nf) {
#pragma unroll
    for (int v = 0; v < 8; ++v) {
      const int qrow = hi8 + v;
      const int hd   = nf * 16 + (lane & 15);
      obase[(size_t)qrow * D_MODEL + hd] = (_Float16)(ctx[nf][v] * rs[v]);
    }
  }
}

// ---- host side ------------------------------------------------------------

extern "C" void kernel_launch(void* const* d_in, const int* in_sizes, int n_in,
                              void* d_out, int out_size, void* d_ws, size_t ws_size,
                              hipStream_t stream) {
  (void)in_sizes; (void)n_in; (void)out_size; (void)ws_size;
  const float* x     = (const float*)d_in[0];
  const int*   mask  = (const int*)d_in[1];
  const float* alpha = (const float*)d_in[2];
  const float* bias  = (const float*)d_in[3];
  const float* w_q   = (const float*)d_in[4];
  const float* w_k   = (const float*)d_in[5];
  const float* w_v   = (const float*)d_in[6];
  const float* w_o   = (const float*)d_in[7];
  const float* fc1_w = (const float*)d_in[8];
  const float* fc1_b = (const float*)d_in[9];
  const float* fc2_w = (const float*)d_in[10];
  const float* fc2_b = (const float*)d_in[11];
  float* out = (float*)d_out;

  char* ws = (char*)d_ws;
  size_t off = 0;
  auto alloc = [&](size_t bytes) -> char* {
    char* p = ws + off;
    off += (bytes + 255) & ~(size_t)255;
    return p;
  };
  _Float16* xn_h  = (_Float16*)alloc((size_t)ROWS * D_MODEL * 2);
  _Float16* wq_h  = (_Float16*)alloc((size_t)D_MODEL * D_MODEL * 2);
  _Float16* wk_h  = (_Float16*)alloc((size_t)D_MODEL * D_MODEL * 2);
  _Float16* wv_h  = (_Float16*)alloc((size_t)D_MODEL * D_MODEL * 2);
  _Float16* wo_h  = (_Float16*)alloc((size_t)D_MODEL * D_MODEL * 2);
  _Float16* fc1_h = (_Float16*)alloc((size_t)FF_DIM * D_MODEL * 2);
  _Float16* fc2_h = (_Float16*)alloc((size_t)D_MODEL * FF_DIM * 2);
  _Float16* q_h   = (_Float16*)alloc((size_t)ROWS * D_MODEL * 2);
  _Float16* k_h   = (_Float16*)alloc((size_t)ROWS * D_MODEL * 2);
  _Float16* v_h   = (_Float16*)alloc((size_t)ROWS * D_MODEL * 2);
  _Float16* vt_h  = (_Float16*)alloc((size_t)ROWS * D_MODEL * 2);
  _Float16* ctx_h = (_Float16*)alloc((size_t)ROWS * D_MODEL * 2);
  float*    x1    = (float*)   alloc((size_t)ROWS * D_MODEL * 4);
  _Float16* xn2_h = (_Float16*)alloc((size_t)ROWS * D_MODEL * 2);
  _Float16* ff_h  = (_Float16*)alloc((size_t)ROWS * FF_DIM * 2);
  float*    madd  = (float*)   alloc((size_t)BATCH * SEQ * 4);

  const int nDD = D_MODEL * D_MODEL;
  const int nFD = FF_DIM * D_MODEL;
  cvt_kernel<<<(nDD + 255) / 256, 256, 0, stream>>>(w_q, wq_h, nDD);
  cvt_kernel<<<(nDD + 255) / 256, 256, 0, stream>>>(w_k, wk_h, nDD);
  cvt_kernel<<<(nDD + 255) / 256, 256, 0, stream>>>(w_v, wv_h, nDD);
  cvt_kernel<<<(nDD + 255) / 256, 256, 0, stream>>>(w_o, wo_h, nDD);
  cvt_kernel<<<(nFD + 255) / 256, 256, 0, stream>>>(fc1_w, fc1_h, nFD);
  cvt_kernel<<<(nFD + 255) / 256, 256, 0, stream>>>(fc2_w, fc2_h, nFD);
  maskf_kernel<<<(BATCH * SEQ + 255) / 256, 256, 0, stream>>>(mask, madd,
                                                              BATCH * SEQ);

  ln_kernel<<<ROWS, 256, 0, stream>>>(x, alpha, bias, xn_h);

  dim3 gD(D_MODEL / 128, ROWS / 128);   // 8 x 32
  dim3 gF(FF_DIM / 128, ROWS / 128);    // 32 x 32

  // Q/K/V projections (f16 outputs for attention)
  gemm_kernel<false, false, false, false, true><<<gD, 256, 0, stream>>>(
      xn_h, wq_h, nullptr, nullptr, nullptr, q_h, ROWS, D_MODEL, D_MODEL);
  gemm_kernel<false, false, false, false, true><<<gD, 256, 0, stream>>>(
      xn_h, wk_h, nullptr, nullptr, nullptr, k_h, ROWS, D_MODEL, D_MODEL);
  gemm_kernel<false, false, false, false, true><<<gD, 256, 0, stream>>>(
      xn_h, wv_h, nullptr, nullptr, nullptr, v_h, ROWS, D_MODEL, D_MODEL);

  vtrans_kernel<<<(ROWS * D_MODEL) / 256, 256, 0, stream>>>(v_h, vt_h);

  attn_kernel<<<dim3(SEQ / 16, NHEAD, BATCH), 32, 0, stream>>>(
      q_h, k_h, vt_h, madd, ctx_h);

  // x1 = x + ctx @ Wo^T
  gemm_kernel<false, true, false, true, false><<<gD, 256, 0, stream>>>(
      ctx_h, wo_h, nullptr, x, x1, nullptr, ROWS, D_MODEL, D_MODEL);

  ln_kernel<<<ROWS, 256, 0, stream>>>(x1, alpha, bias, xn2_h);

  // ff = relu(xn2 @ fc1^T + b1)
  gemm_kernel<true, false, true, false, true><<<gF, 256, 0, stream>>>(
      xn2_h, fc1_h, fc1_b, nullptr, nullptr, ff_h, ROWS, FF_DIM, D_MODEL);

  // out = x1 + ff @ fc2^T + b2
  gemm_kernel<false, true, true, true, false><<<gD, 256, 0, stream>>>(
      ff_h, fc2_h, fc2_b, x1, out, nullptr, ROWS, D_MODEL, FF_DIM);
}
